// MultiheadAttention_57629871177835
// MI455X (gfx1250) — matrix operand order repaired and checked by reading, not compiled, not run on hardware
//
#include <hip/hip_runtime.h>
#include <hip/hip_bf16.h>

typedef __bf16 bf16_t;
typedef __attribute__((ext_vector_type(16))) __bf16 bf16x16;
typedef __attribute__((ext_vector_type(8)))  __bf16 bf16x8;
typedef __attribute__((ext_vector_type(8)))  float  f32x8;

#define NH 8
#define DH 128
#define NB 8
#define S_ 1024
#define DM 1024            // NH*DH
#define MR (NB*S_)         // 8192 rows

static __device__ __forceinline__ f32x8 wmma_bf16(bf16x16 a, bf16x16 b, f32x8 c) {
  // D = A(16x32) * B(32x16) + C, fp32 accumulate
  return __builtin_amdgcn_wmma_f32_16x16x32_bf16(false, a, false, b, (short)0, c, false, false);
}

union ABu { bf16x16 v; bf16x8 h[2]; };

// A-operand: lane holds row l%16; elements k = (i<8?i:i+8) + 8*(lane>=16)
static __device__ __forceinline__ bf16x16 load_A(const bf16_t* p, int base0) {
  ABu u;
  u.h[0] = *(const bf16x8*)(p + base0);
  u.h[1] = *(const bf16x8*)(p + base0 + 16);
  return u.v;
}
// B-operand: lane holds col l%16; elements k = (i<8?i:i+8) + 8*(lane>=16)
static __device__ __forceinline__ bf16x16 load_B(const bf16_t* p, int hi) {
  ABu u;
  const bf16_t* q = p + hi * 8;
  u.h[0] = *(const bf16x8*)(q);
  u.h[1] = *(const bf16x8*)(q + 16);
  return u.v;
}
// A-operand sourced from fp32 memory (convert to bf16 on load)
static __device__ __forceinline__ bf16x16 load_A_f32(const float* p, int base0) {
  bf16x16 r;
#pragma unroll
  for (int i = 0; i < 8; ++i) r[i] = (bf16_t)p[base0 + i];
#pragma unroll
  for (int i = 0; i < 8; ++i) r[i + 8] = (bf16_t)p[base0 + 16 + i];
  return r;
}

// ---------------- Stage 0: weight transpose + bf16 convert --------------------
__global__ __launch_bounds__(256) void wtrans_kernel(const float* __restrict__ W,
                                                     bf16_t* __restrict__ Wt,
                                                     int R, int C) {
  int i = blockIdx.x * blockDim.x + threadIdx.x;
  if (i >= R * C) return;
  int r = i / C, c = i - r * C;
  Wt[(size_t)c * R + r] = (bf16_t)W[i];   // Wt[out][in]
}

// ---------------- Stage 1: fused QKV projection GEMM --------------------------
// X [MR,128] fp32 @ W [128,1024] + bias -> bf16, stored [b,h,s,d] (vmode=0)
// or transposed [b,h,d,s] (vmode=1). Wave computes a 16x128 output strip.
__global__ __launch_bounds__(128) void qkv_proj_kernel(const float* __restrict__ X,
    const bf16_t* __restrict__ Wt, const float* __restrict__ bias,
    bf16_t* __restrict__ out, int vmode, float scale) {
  const int lane = threadIdx.x & 31, wave = threadIdx.x >> 5;
  const int l16 = lane & 15, hi = lane >> 4, base0 = hi * 8;
  const int mbase = (blockIdx.x * 4 + wave) * 16;
  const int colbase = blockIdx.y * 128;

  bf16x16 A[4];
  const float* xrow = X + (size_t)(mbase + l16) * DH;
#pragma unroll
  for (int c = 0; c < 4; ++c) A[c] = load_A_f32(xrow + c * 32, base0);

  f32x8 acc[8];
#pragma unroll
  for (int nt = 0; nt < 8; ++nt) acc[nt] = 0.0f;

#pragma unroll
  for (int nt = 0; nt < 8; ++nt) {
    const bf16_t* wrow = Wt + (size_t)(colbase + nt * 16 + l16) * DH;
#pragma unroll
    for (int c = 0; c < 4; ++c)
      acc[nt] = wmma_bf16(A[c], load_B(wrow + c * 32, hi), acc[nt]);
  }

#pragma unroll
  for (int nt = 0; nt < 8; ++nt) {
    int col = colbase + nt * 16 + l16;
    float bv = bias[col];
    int h = col >> 7, dcol = col & 127;
#pragma unroll
    for (int v = 0; v < 8; ++v) {
      int r = mbase + v + 8 * hi;
      int b = r >> 10, s = r & 1023;
      float val = (acc[nt][v] + bv) * scale;
      size_t idx = (vmode == 0)
          ? (((size_t)(b * NH + h) * S_ + s) * DH + dcol)
          : (((size_t)(b * NH + h) * DH + dcol) * S_ + s);
      out[idx] = (bf16_t)val;
    }
  }
}

// ---------------- Stage 2: causal flash attention -----------------------------
// grid (S/64, B*NH); 4 waves/block, wave = one 16-row Q strip (Q pre-scaled).
__global__ __launch_bounds__(128) void attn_kernel(const bf16_t* __restrict__ Qb,
    const bf16_t* __restrict__ Kb, const bf16_t* __restrict__ Vt,
    bf16_t* __restrict__ Ob) {
  __shared__ bf16_t sP[4][16 * 32];   // per-wave P staging (D-layout -> A-layout)
  const int lane = threadIdx.x & 31, wave = threadIdx.x >> 5;
  const int l16 = lane & 15, hi = lane >> 4, base0 = hi * 8;
  const int bh = blockIdx.y, b = bh >> 3, h = bh & 7;
  const int qbase = blockIdx.x * 64 + wave * 16;
  const bf16_t* Q = Qb + (size_t)bh * S_ * DH;
  const bf16_t* K = Kb + (size_t)bh * S_ * DH;
  const bf16_t* V = Vt + (size_t)bh * DH * S_;   // [d][s]
  bf16_t* lds = sP[wave];

  bf16x16 qA[4];
  const bf16_t* qrow = Q + (size_t)(qbase + l16) * DH;
#pragma unroll
  for (int c = 0; c < 4; ++c) qA[c] = load_A(qrow + c * 32, base0);

  f32x8 o[8];
#pragma unroll
  for (int t = 0; t < 8; ++t) o[t] = 0.0f;
  float rmax[8], rsum[8];
#pragma unroll
  for (int v = 0; v < 8; ++v) { rmax[v] = -__builtin_inff(); rsum[v] = 0.0f; }

  const int ntiles = (qbase + 15) / 32 + 1;   // causal: keys <= qbase+15
  for (int kt = 0; kt < ntiles; ++kt) {
    const int k0 = kt * 32;
    f32x8 s0 = 0.0f, s1 = 0.0f;
    const bf16_t* kr0 = K + (size_t)(k0 + l16) * DH;
    const bf16_t* kr1 = K + (size_t)(k0 + 16 + l16) * DH;
#pragma unroll
    for (int c = 0; c < 4; ++c) {
      s0 = wmma_bf16(qA[c], load_B(kr0 + c * 32, hi), s0);
      s1 = wmma_bf16(qA[c], load_B(kr1 + c * 32, hi), s1);
    }
    if (k0 + 31 > qbase) {   // boundary tile: mask key > row
#pragma unroll
      for (int v = 0; v < 8; ++v) {
        int row = qbase + v + 8 * hi;
        if (k0 + l16 > row)      s0[v] = -__builtin_inff();
        if (k0 + 16 + l16 > row) s1[v] = -__builtin_inff();
      }
    }
    // online softmax: row m = v + 8*hi lives in one 16-lane half; reduce over lanes
#pragma unroll
    for (int v = 0; v < 8; ++v) {
      float lm = fmaxf(s0[v], s1[v]);
      lm = fmaxf(lm, __shfl_xor(lm, 1));
      lm = fmaxf(lm, __shfl_xor(lm, 2));
      lm = fmaxf(lm, __shfl_xor(lm, 4));
      lm = fmaxf(lm, __shfl_xor(lm, 8));
      float nm = fmaxf(rmax[v], lm);
      float corr = exp2f((rmax[v] - nm) * 1.44269504089f);
      rmax[v] = nm;
      float p0 = exp2f((s0[v] - nm) * 1.44269504089f);
      float p1 = exp2f((s1[v] - nm) * 1.44269504089f);
      float ps = p0 + p1;
      ps += __shfl_xor(ps, 1);
      ps += __shfl_xor(ps, 2);
      ps += __shfl_xor(ps, 4);
      ps += __shfl_xor(ps, 8);
      rsum[v] = rsum[v] * corr + ps;
#pragma unroll
      for (int t = 0; t < 8; ++t) o[t][v] *= corr;
      int m = v + 8 * hi;
      lds[m * 32 + l16]      = (bf16_t)p0;   // P[m][key] staging
      lds[m * 32 + 16 + l16] = (bf16_t)p1;
    }
    asm volatile("s_wait_dscnt 0x0" ::: "memory");   // LDS in-order per wave
    bf16x16 pA = load_A(lds + l16 * 32, base0);
    asm volatile("" ::: "memory");
    // O += P(16x32) * V(32x128): V^T layout makes B contiguous along keys
#pragma unroll
    for (int t = 0; t < 8; ++t) {
      const bf16_t* vr = V + (size_t)(t * 16 + l16) * S_ + k0;
      o[t] = wmma_bf16(pA, load_B(vr, hi), o[t]);
    }
  }

#pragma unroll
  for (int v = 0; v < 8; ++v) {
    float inv = 1.0f / rsum[v];
    int srow = qbase + v + 8 * hi;
#pragma unroll
    for (int t = 0; t < 8; ++t) {
      size_t idx = ((size_t)b * S_ + srow) * DM + h * DH + t * 16 + l16;
      Ob[idx] = (bf16_t)(o[t][v] * inv);
    }
  }
}

// ---------------- Stage 3: output projection ----------------------------------
// Ob [MR,1024] bf16 @ Wo [1024,128] + bo -> fp32 out. Wave = 16x128 strip.
__global__ __launch_bounds__(128) void out_proj_kernel(const bf16_t* __restrict__ Ob,
    const bf16_t* __restrict__ Wot, const float* __restrict__ bo,
    float* __restrict__ out) {
  const int lane = threadIdx.x & 31, wave = threadIdx.x >> 5;
  const int l16 = lane & 15, hi = lane >> 4, base0 = hi * 8;
  const int mbase = (blockIdx.x * 4 + wave) * 16;
  f32x8 acc[8];
#pragma unroll
  for (int nt = 0; nt < 8; ++nt) acc[nt] = 0.0f;
  const bf16_t* arow = Ob + (size_t)(mbase + l16) * DM;
  for (int c = 0; c < 32; ++c) {
    bf16x16 A = load_A(arow + c * 32, base0);
#pragma unroll
    for (int nt = 0; nt < 8; ++nt) {
      const bf16_t* wrow = Wot + (size_t)(nt * 16 + l16) * DM + c * 32;
      acc[nt] = wmma_bf16(A, load_B(wrow, hi), acc[nt]);
    }
  }
#pragma unroll
  for (int nt = 0; nt < 8; ++nt) {
    int col = nt * 16 + l16;
    float bv = bo[col];
#pragma unroll
    for (int v = 0; v < 8; ++v) {
      int r = mbase + v + 8 * hi;
      out[(size_t)r * DH + col] = acc[nt][v] + bv;
    }
  }
}

extern "C" void kernel_launch(void* const* d_in, const int* in_sizes, int n_in,
                              void* d_out, int out_size, void* d_ws, size_t ws_size,
                              hipStream_t stream) {
  const float* q  = (const float*)d_in[0];
  const float* k  = (const float*)d_in[1];
  const float* v  = (const float*)d_in[2];
  const float* Wq = (const float*)d_in[3];
  const float* bq = (const float*)d_in[4];
  const float* Wk = (const float*)d_in[5];
  const float* bk = (const float*)d_in[6];
  const float* Wv = (const float*)d_in[7];
  const float* bv = (const float*)d_in[8];
  const float* Wo = (const float*)d_in[9];
  const float* bo = (const float*)d_in[10];
  float* out = (float*)d_out;

  char* ws = (char*)d_ws;
  bf16_t* WtQ = (bf16_t*)(ws);                               // 1024x128 bf16
  bf16_t* WtK = (bf16_t*)(ws + (256 << 10));
  bf16_t* WtV = (bf16_t*)(ws + (512 << 10));
  bf16_t* WoT = (bf16_t*)(ws + (768 << 10));                 // 128x1024 bf16
  bf16_t* Qb  = (bf16_t*)(ws + (1 << 20));                   // [b,h,s,d] 16MB
  bf16_t* Kb  = (bf16_t*)(ws + (1 << 20) + (16u << 20));     // [b,h,s,d] 16MB
  bf16_t* Vt  = (bf16_t*)(ws + (1 << 20) + (32u << 20));     // [b,h,d,s] 16MB
  bf16_t* Ob  = (bf16_t*)(ws + (1 << 20) + (48u << 20));     // [b,s,D]   16MB

  wtrans_kernel<<<512, 256, 0, stream>>>(Wq, WtQ, 128, 1024);
  wtrans_kernel<<<512, 256, 0, stream>>>(Wk, WtK, 128, 1024);
  wtrans_kernel<<<512, 256, 0, stream>>>(Wv, WtV, 128, 1024);
  wtrans_kernel<<<512, 256, 0, stream>>>(Wo, WoT, 1024, 128);

  const float qscale = 0.08838834764831845f;  // 1/sqrt(128), folded into Q
  dim3 g1(128, 8);
  qkv_proj_kernel<<<g1, 128, 0, stream>>>(q, WtQ, bq, Qb, 0, qscale);
  qkv_proj_kernel<<<g1, 128, 0, stream>>>(k, WtK, bk, Kb, 0, 1.0f);
  qkv_proj_kernel<<<g1, 128, 0, stream>>>(v, WtV, bv, Vt, 1, 1.0f);

  attn_kernel<<<dim3(16, 64), 128, 0, stream>>>(Qb, Kb, Vt, Ob);
  out_proj_kernel<<<128, 128, 0, stream>>>(Ob, WoT, bo, out);

  (void)in_sizes; (void)n_in; (void)out_size; (void)ws_size;
}
